// LSTM_28819230556619
// MI455X (gfx1250) — matrix operand order repaired and checked by reading, not compiled, not run on hardware
//
#include <hip/hip_runtime.h>
#include <hip/hip_bf16.h>

typedef _Float16 v16h __attribute__((ext_vector_type(16)));
typedef _Float16 v8h  __attribute__((ext_vector_type(8)));
typedef _Float16 v2h  __attribute__((ext_vector_type(2)));
typedef float    v8f  __attribute__((ext_vector_type(8)));

// Problem dims: B=32, T=32, IN=128, H=256, A=32, 2H=512
#define LDA       520                    // padded f16 row stride (kills LDS bank conflicts)
#define FEAT_B    (32 * LDA * 2)         // 33280 bytes
#define SMEM_BYTES (FEAT_B * 2 + 32768 + 1024 + 1024 + 1536 + 128 + 128)  // 103168

// workspace layout (bytes); W1s/W2s are fragment-major swizzled f16 weights
#define WS_W1S   0                      // 512x512 f16 swizzled
#define WS_W2S   (512*512*2)            // 512x512 f16 swizzled
#define WS_WIHT  (WS_W2S + 512*512*2)   // 384x1024 f32 (transposed W_ih)
#define WS_WHHT  (WS_WIHT + 384*1024*4) // 256x1024 f32 (transposed W_hh)

// Swizzled index: chunk = idx>>4 (32B chunks of 16 f16), within chunk e = idx&15.
// chunk bits: half(1) | m16(4) | j(3) | kt(4) | g(2)  [LSB..MSB]
// source: n = g*128 + j*16 + m16 ; k = kt*32 + half*16 + e ; value = W[n*512 + k]
__global__ void prep_kernel(const float* __restrict__ W1, const float* __restrict__ W2,
                            const float* __restrict__ Wih, const float* __restrict__ Whh,
                            _Float16* __restrict__ W1s, _Float16* __restrict__ W2s,
                            float* __restrict__ WihT, float* __restrict__ WhhT) {
  int idx = blockIdx.x * 256 + threadIdx.x;
  if (idx < 512 * 512) {
    int e    = idx & 15;
    int chk  = idx >> 4;
    int half = chk & 1;
    int m16  = (chk >> 1) & 15;
    int j    = (chk >> 5) & 7;
    int kt   = (chk >> 8) & 15;
    int g    = (chk >> 12) & 3;
    int src  = (g * 128 + j * 16 + m16) * 512 + kt * 32 + half * 16 + e;
    W1s[idx] = (_Float16)W1[src];
    W2s[idx] = (_Float16)W2[src];
    int o = idx >> 8, jj = idx & 255;           // W_hh is (1024,256)
    WhhT[jj * 1024 + o] = Whh[idx];
  }
  if (idx < 1024 * 384) {                        // W_ih is (1024,384)
    int o = idx / 384, k = idx - o * 384;
    WihT[k * 1024 + o] = Wih[idx];
  }
}

// C(32x512) += A(32x512,f16 LDS, row stride LDA) * W^T (swizzled f16, global).
// Wave owns M-tile mt and 8 N-tiles from nb. Per K-slice: preload all 8 B
// fragments (contiguous 1KB per tile -> fully coalesced b128 pairs), then run
// the 8-wmma chain; partial s_wait_loadcnt keeps loads in flight under wmma.
__device__ __forceinline__ void gemm_acc(const _Float16* __restrict__ Af,
                                         const _Float16* __restrict__ Ws,
                                         v8f acc[8], int mt, int nb, int lane) {
  const int half = lane >> 4;
  const int m16  = lane & 15;
  const _Float16* arow  = Af + (mt * 16 + m16) * LDA + 16 * half;
  const _Float16* bbase = Ws + (nb >> 3) * (16 * 8 * 512) + (m16 * 2 + half) * 16;
  #pragma unroll 4
  for (int kt = 0; kt < 16; ++kt) {
    // A fragment contiguous: K = {16h..16h+15} (matches swizzled B)
    const _Float16* ap = arow + kt * 32;
    v8h a0 = *(const v8h*)(ap);
    v8h a1 = *(const v8h*)(ap + 8);
    v16h av = __builtin_shufflevector(a0, a1, 0,1,2,3,4,5,6,7,8,9,10,11,12,13,14,15);
    // Preload 8 B fragments from the fragment-major swizzled layout
    v16h bv[8];
    #pragma unroll
    for (int j = 0; j < 8; ++j) {
      const _Float16* bp = bbase + kt * (8 * 512) + j * 512;
      v8h b0 = *(const v8h*)(bp);
      v8h b1 = *(const v8h*)(bp + 8);
      bv[j] = __builtin_shufflevector(b0, b1, 0,1,2,3,4,5,6,7,8,9,10,11,12,13,14,15);
    }
    #pragma unroll
    for (int j = 0; j < 8; ++j) {
      acc[j] = __builtin_amdgcn_wmma_f32_16x16x32_f16(
          false, av, false, bv[j], (short)0, acc[j], false, false);
    }
  }
}

__global__ __launch_bounds__(256) void lstm_attn_kernel(
    const float* __restrict__ x, const float* __restrict__ h0,
    const float* __restrict__ c0, const float* __restrict__ out0,
    const float* __restrict__ bih, const float* __restrict__ bhh,
    const float* __restrict__ b1, const float* __restrict__ b2,
    const float* __restrict__ W3, const float* __restrict__ b3,
    const _Float16* __restrict__ W1s, const _Float16* __restrict__ W2s,
    const float* __restrict__ WihT, const float* __restrict__ WhhT,
    float* __restrict__ out) {
  extern __shared__ char smem[];
  _Float16* featA = (_Float16*)(smem);                // 32 x LDA f16
  _Float16* f1h   = (_Float16*)(smem + FEAT_B);       // 32 x LDA f16
  float* buf32    = (float*)(smem + 2 * FEAT_B);      // 32 x 256 f32
  float* hs       = (float*)(smem + 2 * FEAT_B + 32768);
  float* cs       = hs + 256;
  float* xin      = cs + 256;                          // 384
  float* scores   = xin + 384;                         // 32
  float* wvec     = scores + 32;                       // 32

  const int tid  = threadIdx.x;
  const int b    = blockIdx.x;
  const int lane = tid & 31;
  const int wv   = tid >> 5;
  const int mt   = wv & 1;
  const int nb   = (wv >> 1) * 8;
  const int half = lane >> 4;
  const int m16  = lane & 15;

  // ---- init state ----
  #pragma unroll
  for (int a = 0; a < 32; ++a) buf32[a * 256 + tid] = 0.f;
  hs[tid] = h0[b * 256 + tid];
  cs[tid] = c0[b * 256 + tid];
  buf32[tid] = out0[b * 256 + tid];               // buf[:,0,:] = out
  const float bi0 = bih[tid]       + bhh[tid];
  const float bi1 = bih[tid + 256] + bhh[tid + 256];
  const float bi2 = bih[tid + 512] + bhh[tid + 512];
  const float bi3 = bih[tid + 768] + bhh[tid + 768];
  const float b3v = b3[0];
  const float* xb = x + b * 32 * 128;
  __syncthreads();

  #pragma unroll 1
  for (int t = 0; t < 32; ++t) {
    // 1. feat = [buf | broadcast(h)] as f16 (packed 32-bit stores); scores = b3
    #pragma unroll 4
    for (int idx = tid; idx < 32 * 128; idx += 256) {
      int a = idx >> 7, j2 = (idx & 127) << 1;
      v2h pb = { (_Float16)buf32[a * 256 + j2], (_Float16)buf32[a * 256 + j2 + 1] };
      *(v2h*)(featA + a * LDA + j2) = pb;
      v2h ph = { (_Float16)hs[j2], (_Float16)hs[j2 + 1] };
      *(v2h*)(featA + a * LDA + 256 + j2) = ph;
    }
    if (tid < 32) scores[tid] = b3v;
    __syncthreads();

    // 2. layer1: f1 = relu(feat @ W1^T + b1) -> f16 LDS
    {
      v8f acc[8];
      #pragma unroll
      for (int j = 0; j < 8; ++j) {
        #pragma unroll
        for (int r = 0; r < 8; ++r) acc[j][r] = 0.f;
      }
      gemm_acc(featA, W1s, acc, mt, nb, lane);
      #pragma unroll
      for (int j = 0; j < 8; ++j) {
        const int col = (nb + j) * 16 + m16;
        const float bias = b1[col];
        #pragma unroll
        for (int r = 0; r < 8; ++r) {
          const int row = mt * 16 + r + 8 * half;   // C layout: VGPR r, lane halves
          float v = acc[j][r] + bias;
          f1h[row * LDA + col] = (_Float16)(v > 0.f ? v : 0.f);
        }
      }
    }
    __syncthreads();

    // 3. layer2 + layer3 fused: scores[row] += sum_col relu(f1@W2^T+b2)[row,col]*W3[col]
    {
      v8f acc[8];
      #pragma unroll
      for (int j = 0; j < 8; ++j) {
        #pragma unroll
        for (int r = 0; r < 8; ++r) acc[j][r] = 0.f;
      }
      gemm_acc(f1h, W2s, acc, mt, nb, lane);
      float sp[8];
      #pragma unroll
      for (int r = 0; r < 8; ++r) sp[r] = 0.f;
      #pragma unroll
      for (int j = 0; j < 8; ++j) {
        const int col = (nb + j) * 16 + m16;
        const float bias = b2[col];
        const float w3v = W3[col];
        #pragma unroll
        for (int r = 0; r < 8; ++r) {
          float v = acc[j][r] + bias;
          sp[r] += (v > 0.f ? v : 0.f) * w3v;
        }
      }
      #pragma unroll
      for (int r = 0; r < 8; ++r)
        atomicAdd(&scores[mt * 16 + r + 8 * half], sp[r]);
    }
    __syncthreads();

    // 4. masked softmax over A=32 (valid: a <= t); tiny, serial in one lane
    if (tid == 0) {
      float mx = -1e30f;
      for (int a = 0; a < 32; ++a) {
        float s = (a <= t) ? scores[a] : -1.0e9f;
        scores[a] = s;
        mx = fmaxf(mx, s);
      }
      float sum = 0.f;
      for (int a = 0; a < 32; ++a) { float e = __expf(scores[a] - mx); wvec[a] = e; sum += e; }
      float inv = 1.f / sum;
      for (int a = 0; a < 32; ++a) wvec[a] *= inv;
    }
    __syncthreads();

    // 5. attention (f32 from buf32) + xin = [x_t | atten]
    {
      float at = 0.f;
      #pragma unroll 4
      for (int a = 0; a < 32; ++a) at += wvec[a] * buf32[a * 256 + tid];
      xin[128 + tid] = at;
      if (tid < 128) xin[tid] = xb[t * 128 + tid];
    }
    __syncthreads();

    // 6. gates GEMV with transposed (coalesced) f32 weights
    float g0 = bi0, g1 = bi1, g2 = bi2, g3 = bi3;
    #pragma unroll 4
    for (int k = 0; k < 384; ++k) {
      const float xv = xin[k];
      const float* p = WihT + k * 1024 + tid;
      g0 += p[0] * xv; g1 += p[256] * xv; g2 += p[512] * xv; g3 += p[768] * xv;
    }
    #pragma unroll 4
    for (int j = 0; j < 256; ++j) {
      const float hv = hs[j];
      const float* p = WhhT + j * 1024 + tid;
      g0 += p[0] * hv; g1 += p[256] * hv; g2 += p[512] * hv; g3 += p[768] * hv;
    }
    __syncthreads();

    // 7. LSTM update (f32)
    const float ig = 1.f / (1.f + __expf(-g0));
    const float fg = 1.f / (1.f + __expf(-g1));
    const float gg = tanhf(g2);
    const float og = 1.f / (1.f + __expf(-g3));
    const float cn = fg * cs[tid] + ig * gg;
    const float hn = og * tanhf(cn);
    cs[tid] = cn;
    hs[tid] = hn;
    buf32[((t + 1) & 31) * 256 + tid] = hn;
    __syncthreads();
  }

  // ---- outputs: roll(buf,-1) , h , c ----
  #pragma unroll 4
  for (int idx = tid; idx < 32 * 256; idx += 256) {
    int a = idx >> 8, j = idx & 255;
    out[(b * 32 + a) * 256 + j] = buf32[(((a + 1) & 31) << 8) + j];
  }
  out[32 * 32 * 256 + b * 256 + tid]            = hs[tid];
  out[32 * 32 * 256 + 32 * 256 + b * 256 + tid] = cs[tid];
}

extern "C" void kernel_launch(void* const* d_in, const int* in_sizes, int n_in,
                              void* d_out, int out_size, void* d_ws, size_t ws_size,
                              hipStream_t stream) {
  const float* x    = (const float*)d_in[0];
  const float* h0   = (const float*)d_in[1];
  const float* c0   = (const float*)d_in[2];
  const float* out0 = (const float*)d_in[3];
  const float* Wih  = (const float*)d_in[4];
  const float* Whh  = (const float*)d_in[5];
  const float* bih  = (const float*)d_in[6];
  const float* bhh  = (const float*)d_in[7];
  const float* W1   = (const float*)d_in[8];
  const float* b1   = (const float*)d_in[9];
  const float* W2   = (const float*)d_in[10];
  const float* b2   = (const float*)d_in[11];
  const float* W3   = (const float*)d_in[12];
  const float* b3   = (const float*)d_in[13];

  char* ws = (char*)d_ws;
  _Float16* W1s  = (_Float16*)(ws + WS_W1S);
  _Float16* W2s  = (_Float16*)(ws + WS_W2S);
  float*    WihT = (float*)(ws + WS_WIHT);
  float*    WhhT = (float*)(ws + WS_WHHT);

  (void)hipFuncSetAttribute(reinterpret_cast<const void*>(lstm_attn_kernel),
                            hipFuncAttributeMaxDynamicSharedMemorySize, SMEM_BYTES);

  prep_kernel<<<(1024 * 384 + 255) / 256, 256, 0, stream>>>(
      W1, W2, Wih, Whh, W1s, W2s, WihT, WhhT);

  lstm_attn_kernel<<<32, 256, SMEM_BYTES, stream>>>(
      x, h0, c0, out0, bih, bhh, b1, b2, W3, b3, W1s, W2s, WihT, WhhT,
      (float*)d_out);
}
